// RuntimePredictorLSTM_80401787781407
// MI455X (gfx1250) — matrix operand order repaired and checked by reading, not compiled, not run on hardware
//
#include <hip/hip_runtime.h>
#include <hip/hip_bf16.h>
#include <math.h>

// ---------------------------------------------------------------------------
// 2-layer LSTM (B=256, T=512, I=64, H=512) + LayerNorm + FC, for MI455X.
// Strategy: bf16 WMMA (v_wmma_f32_16x16x32_bf16) for all GEMMs, f32 state.
// One kernel per (layer, timestep) -> sequential dependency handled by the
// stream/graph; ping-pong h buffers avoid intra-kernel races. Everything
// (weights, states) lives in L2 after the first touch.
// ---------------------------------------------------------------------------

typedef __attribute__((ext_vector_type(16))) __bf16 v16bf;
typedef __attribute__((ext_vector_type(8)))  float  v8f;

union Frag { v16bf v; uint4 u[2]; };

#define BSZ 256
#define TT  512
#define II  64
#define HH  512
#define BB  32   // batch tile per workgroup
#define BJ  32   // hidden-unit tile per workgroup (x4 gates = 128 gate cols)

// ---------------------------------------------------------------------------
// f32 -> bf16 conversion (grid-stride)
// ---------------------------------------------------------------------------
__global__ void cvt_f32_bf16(const float* __restrict__ src,
                             __bf16* __restrict__ dst, size_t n) {
  size_t i = (size_t)blockIdx.x * blockDim.x + threadIdx.x;
  size_t stride = (size_t)gridDim.x * blockDim.x;
  for (; i < n; i += stride) dst[i] = (__bf16)src[i];
}

// ---------------------------------------------------------------------------
// One LSTM step for one layer.
// grid = (B/BB, H/BJ), block = 128 (4 waves; wave w handles gate w).
// Each wave computes a 32x32 preactivation tile for its gate with WMMA,
// gates meet in LDS, then the cell update runs fused.
// ---------------------------------------------------------------------------
__global__ __launch_bounds__(128)
void lstm_step(const __bf16* __restrict__ xin, long xin_stride, int Kin,
               const __bf16* __restrict__ hprev,          // B x H (bf16)
               const __bf16* __restrict__ wi,             // (4H x Kin) bf16 row-major
               const __bf16* __restrict__ wh,             // (4H x H)   bf16 row-major
               const float*  __restrict__ b_ih,
               const float*  __restrict__ b_hh,
               float*  __restrict__ c,                    // B x H f32, in/out
               __bf16* __restrict__ hout,                 // B x H bf16
               float*  __restrict__ hout_f32,             // B x H f32
               int first)
{
  __shared__ float preact[BB * 4 * BJ];                   // 32 x 128 f32 = 16 KB

  const int lane   = threadIdx.x & 31;
  const int g      = threadIdx.x >> 5;                    // gate 0..3 (i,f,g,o)
  const int laneLo = lane & 15;
  const int hiHalf = lane >> 4;                           // 0 or 1
  const int bBase  = blockIdx.x * BB;
  const int jBase  = blockIdx.y * BJ;

  v8f acc[2][2] = {};

  // ---- input GEMM: xg += x_t @ wi^T (tile columns g*H + jBase .. +31) ----
  {
    const __bf16* wrow = wi + (size_t)(g * HH + jBase) * Kin;
    for (int k0 = 0; k0 < Kin; k0 += 32) {
      Frag a[2], b[2];
#pragma unroll
      for (int mi = 0; mi < 2; ++mi) {
        const __bf16* p = xin + (size_t)(bBase + mi * 16 + laneLo) * xin_stride
                              + k0 + hiHalf * 8;
        a[mi].u[0] = *(const uint4*)p;
        a[mi].u[1] = *(const uint4*)(p + 16);
      }
#pragma unroll
      for (int ni = 0; ni < 2; ++ni) {
        const __bf16* p = wrow + (size_t)(ni * 16 + laneLo) * Kin
                               + k0 + hiHalf * 8;
        b[ni].u[0] = *(const uint4*)p;
        b[ni].u[1] = *(const uint4*)(p + 16);
      }
#pragma unroll
      for (int mi = 0; mi < 2; ++mi)
#pragma unroll
        for (int ni = 0; ni < 2; ++ni)
          acc[mi][ni] = __builtin_amdgcn_wmma_f32_16x16x32_bf16(
              false, a[mi].v, false, b[ni].v, (short)0, acc[mi][ni],
              false, false);
    }
  }

  // ---- recurrent GEMM: += h_{t-1} @ wh^T (skipped at t==0, h0 == 0) ----
  if (!first) {
    const __bf16* wrow = wh + (size_t)(g * HH + jBase) * HH;
    for (int k0 = 0; k0 < HH; k0 += 32) {
      // weights stream straight from L2; hint the next block
      __builtin_prefetch(wrow + (size_t)laneLo * HH + k0 + 32, 0, 1);
      Frag a[2], b[2];
#pragma unroll
      for (int mi = 0; mi < 2; ++mi) {
        const __bf16* p = hprev + (size_t)(bBase + mi * 16 + laneLo) * HH
                                + k0 + hiHalf * 8;
        a[mi].u[0] = *(const uint4*)p;
        a[mi].u[1] = *(const uint4*)(p + 16);
      }
#pragma unroll
      for (int ni = 0; ni < 2; ++ni) {
        const __bf16* p = wrow + (size_t)(ni * 16 + laneLo) * HH
                               + k0 + hiHalf * 8;
        b[ni].u[0] = *(const uint4*)p;
        b[ni].u[1] = *(const uint4*)(p + 16);
      }
#pragma unroll
      for (int mi = 0; mi < 2; ++mi)
#pragma unroll
        for (int ni = 0; ni < 2; ++ni)
          acc[mi][ni] = __builtin_amdgcn_wmma_f32_16x16x32_bf16(
              false, a[mi].v, false, b[ni].v, (short)0, acc[mi][ni],
              false, false);
    }
  }

  // ---- spill preactivations for this gate into LDS ----
  // D layout: lane L, vgpr v -> row M = v + (L>=16)*8, col N = L%16
#pragma unroll
  for (int mi = 0; mi < 2; ++mi)
#pragma unroll
    for (int ni = 0; ni < 2; ++ni)
#pragma unroll
      for (int v = 0; v < 8; ++v) {
        int m = mi * 16 + hiHalf * 8 + v;
        int n = ni * 16 + laneLo;
        preact[m * (4 * BJ) + g * BJ + n] = acc[mi][ni][v];
      }
  __syncthreads();

  // ---- fused LSTM cell update: 32x32 cells, 8 per thread ----
  for (int idx = threadIdx.x; idx < BB * BJ; idx += 128) {
    const int m  = idx >> 5;         // local batch row
    const int j  = idx & 31;         // local hidden unit
    const int b  = bBase + m;
    const int jg = jBase + j;
    const size_t sidx = (size_t)b * HH + jg;

    float ig = preact[m * 128 + 0 * BJ + j] + b_ih[0 * HH + jg] + b_hh[0 * HH + jg];
    float fg = preact[m * 128 + 1 * BJ + j] + b_ih[1 * HH + jg] + b_hh[1 * HH + jg];
    float gg = preact[m * 128 + 2 * BJ + j] + b_ih[2 * HH + jg] + b_hh[2 * HH + jg];
    float og = preact[m * 128 + 3 * BJ + j] + b_ih[3 * HH + jg] + b_hh[3 * HH + jg];

    float i_ = 1.0f / (1.0f + __expf(-ig));
    float f_ = 1.0f / (1.0f + __expf(-fg));
    float g_ = tanhf(gg);
    float o_ = 1.0f / (1.0f + __expf(-og));

    float cprev = first ? 0.0f : c[sidx];
    float cn = f_ * cprev + i_ * g_;
    float hn = o_ * tanhf(cn);

    c[sidx]        = cn;
    hout[sidx]     = (__bf16)hn;
    hout_f32[sidx] = hn;
  }
}

// ---------------------------------------------------------------------------
// LayerNorm + FC over the final hidden state. One block per batch row.
// ---------------------------------------------------------------------------
__global__ __launch_bounds__(256)
void ln_fc(const float* __restrict__ hlast,
           const float* __restrict__ ln_g, const float* __restrict__ ln_b,
           const float* __restrict__ fc_w, const float* __restrict__ fc_b,
           float* __restrict__ out)
{
  __shared__ float red[256];
  const int b = blockIdx.x;
  const float* row = hlast + (size_t)b * HH;

  float s = 0.0f;
  for (int j = threadIdx.x; j < HH; j += 256) s += row[j];
  red[threadIdx.x] = s; __syncthreads();
  for (int off = 128; off > 0; off >>= 1) {
    if (threadIdx.x < off) red[threadIdx.x] += red[threadIdx.x + off];
    __syncthreads();
  }
  const float mu = red[0] / HH;
  __syncthreads();

  float v = 0.0f;
  for (int j = threadIdx.x; j < HH; j += 256) {
    float d = row[j] - mu; v += d * d;
  }
  red[threadIdx.x] = v; __syncthreads();
  for (int off = 128; off > 0; off >>= 1) {
    if (threadIdx.x < off) red[threadIdx.x] += red[threadIdx.x + off];
    __syncthreads();
  }
  const float rs = rsqrtf(red[0] / HH + 1e-5f);
  __syncthreads();

  float acc = 0.0f;
  for (int j = threadIdx.x; j < HH; j += 256)
    acc += ((row[j] - mu) * rs * ln_g[j] + ln_b[j]) * fc_w[j];
  red[threadIdx.x] = acc; __syncthreads();
  for (int off = 128; off > 0; off >>= 1) {
    if (threadIdx.x < off) red[threadIdx.x] += red[threadIdx.x + off];
    __syncthreads();
  }
  if (threadIdx.x == 0) out[b] = red[0] + fc_b[0];
}

// ---------------------------------------------------------------------------
extern "C" void kernel_launch(void* const* d_in, const int* in_sizes, int n_in,
                              void* d_out, int out_size, void* d_ws, size_t ws_size,
                              hipStream_t stream) {
  (void)in_sizes; (void)n_in; (void)out_size; (void)ws_size;

  const float* x     = (const float*)d_in[0];
  const float* w_ih0 = (const float*)d_in[1];
  const float* w_hh0 = (const float*)d_in[2];
  const float* b_ih0 = (const float*)d_in[3];
  const float* b_hh0 = (const float*)d_in[4];
  const float* w_ih1 = (const float*)d_in[5];
  const float* w_hh1 = (const float*)d_in[6];
  const float* b_ih1 = (const float*)d_in[7];
  const float* b_hh1 = (const float*)d_in[8];
  const float* ln_g  = (const float*)d_in[9];
  const float* ln_b  = (const float*)d_in[10];
  const float* fc_w  = (const float*)d_in[11];
  const float* fc_b  = (const float*)d_in[12];
  float* out = (float*)d_out;

  // ---- workspace carve-up (all offsets 256B aligned) ----
  char* ws = (char*)d_ws;
  size_t off = 0;
  auto carve = [&](size_t bytes) {
    char* p = ws + off;
    off += (bytes + 255) & ~(size_t)255;
    return p;
  };
  const size_t nX   = (size_t)BSZ * TT * II;          // 8.4M elems
  const size_t nWi0 = (size_t)4 * HH * II;
  const size_t nWh  = (size_t)4 * HH * HH;
  const size_t nBH  = (size_t)BSZ * HH;

  __bf16* xb   = (__bf16*)carve(nX   * 2);
  __bf16* wi0b = (__bf16*)carve(nWi0 * 2);
  __bf16* wh0b = (__bf16*)carve(nWh  * 2);
  __bf16* wi1b = (__bf16*)carve(nWh  * 2);
  __bf16* wh1b = (__bf16*)carve(nWh  * 2);
  __bf16* h0A  = (__bf16*)carve(nBH * 2);
  __bf16* h0B  = (__bf16*)carve(nBH * 2);
  __bf16* h1A  = (__bf16*)carve(nBH * 2);
  __bf16* h1B  = (__bf16*)carve(nBH * 2);
  float*  c0   = (float*) carve(nBH * 4);
  float*  c1   = (float*) carve(nBH * 4);
  float*  h0f  = (float*) carve(nBH * 4);  // layer-0 f32 h (unused sink)
  float*  h1f  = (float*) carve(nBH * 4);  // final-step h of layer 1

  // ---- bf16 conversions (weights + input), recomputed every call ----
  cvt_f32_bf16<<<1024, 256, 0, stream>>>(x,     xb,   nX);
  cvt_f32_bf16<<<256,  256, 0, stream>>>(w_ih0, wi0b, nWi0);
  cvt_f32_bf16<<<512,  256, 0, stream>>>(w_hh0, wh0b, nWh);
  cvt_f32_bf16<<<512,  256, 0, stream>>>(w_ih1, wi1b, nWh);
  cvt_f32_bf16<<<512,  256, 0, stream>>>(w_hh1, wh1b, nWh);

  // ---- sequential scan: fused layer0 -> layer1 per timestep ----
  dim3 grid(BSZ / BB, HH / BJ);   // (8, 16)
  for (int t = 0; t < TT; ++t) {
    const int parity = t & 1;
    __bf16* h0n = parity ? h0B : h0A;
    __bf16* h0p = parity ? h0A : h0B;
    __bf16* h1n = parity ? h1B : h1A;
    __bf16* h1p = parity ? h1A : h1B;

    // layer 0: input row b lives at xb + t*I + b*T*I
    lstm_step<<<grid, 128, 0, stream>>>(
        xb + (size_t)t * II, (long)TT * II, II,
        h0p, wi0b, wh0b, b_ih0, b_hh0, c0, h0n, h0f, t == 0);

    // layer 1: input is layer-0's h_t (contiguous B x H)
    lstm_step<<<grid, 128, 0, stream>>>(
        h0n, (long)HH, HH,
        h1p, wi1b, wh1b, b_ih1, b_hh1, c1, h1n, h1f, t == 0);
  }

  // ---- LayerNorm + FC on the last hidden state ----
  ln_fc<<<BSZ, 256, 0, stream>>>(h1f, ln_g, ln_b, fc_w, fc_b, out);
}
